// Graphormer_EncoderBlock_3556232921559
// MI455X (gfx1250) — matrix operand order repaired and checked by reading, not compiled, not run on hardware
//
#include <hip/hip_runtime.h>
#include <hip/hip_bf16.h>

// ---------------------------------------------------------------------------
// Graphormer encoder block for MI455X (gfx1250, wave32, WMMA + TDM).
// All GEMMs via v_wmma_f32_16x16x32_f16; LDS tiles staged by tensor_load_to_lds.
// ---------------------------------------------------------------------------

typedef __attribute__((ext_vector_type(16))) _Float16     v16h;
typedef __attribute__((ext_vector_type(8)))  _Float16     v8h;
typedef __attribute__((ext_vector_type(8)))  float        v8f;
typedef __attribute__((ext_vector_type(4)))  unsigned int su4;
typedef __attribute__((ext_vector_type(8)))  unsigned int su8;

// ---- WMMA fragment helpers (ISA 7.12.2, 16-bit layouts, wave32) -----------
// A (16xK tile, row-major in LDS): lane L covers row = L%16;
//   halves 0..7  -> k = k0 + (L/16)*8 + i ; halves 8..15 -> +16
// B stored as Bt[n][k] row-major in LDS: lane L covers col n = L%16;
//   halves 0..7  -> k = k0 + (L/16)*8 + i ; halves 8..15 -> +16
__device__ __forceinline__ v16h make_frag(const _Float16* p0, const _Float16* p1) {
    v8h lo = *(const v8h*)p0;
    v8h hi = *(const v8h*)p1;
    v16h r;
#pragma unroll
    for (int i = 0; i < 8; ++i) { r[i] = lo[i]; r[i + 8] = hi[i]; }
    return r;
}
__device__ __forceinline__ v16h frag_a(const _Float16* s, int ld, int lane, int k0) {
    const _Float16* p = s + (lane & 15) * ld + k0 + ((lane >> 4) << 3);
    return make_frag(p, p + 16);
}
__device__ __forceinline__ v16h frag_b(const _Float16* s, int ld, int lane, int k0) {
    const _Float16* p = s + (lane & 15) * ld + k0 + ((lane >> 4) << 3);
    return make_frag(p, p + 16);
}
__device__ __forceinline__ v8f wmma16(v16h a, v16h b, v8f c) {
    return __builtin_amdgcn_wmma_f32_16x16x32_f16(false, a, false, b, (short)0, c,
                                                  false, false);
}

// ---------------------------------------------------------------------------
// Tensor Data Mover: DMA a 2D f16 tile [rows][rowElems] (row stride strideElems)
// from global into LDS at lds_off, with hardware row padding so the LDS layout
// matches the padded pitch used by the WMMA fragment loaders.
// D# layout per ISA 08_async_tensor (group0 128b, group1 256b; groups 2/3 NULL).
//   padInterval: pad after (1<<code) 8-byte units ; padAmount: (code+1) DWORDs.
// Issued by ONE wave (EXEC ignored by TDM); completion via s_wait_tensorcnt.
// ---------------------------------------------------------------------------
__device__ __forceinline__ void tdm_load_tile_f16(unsigned lds_off, const void* gptr,
                                                  unsigned rows, unsigned rowElems,
                                                  unsigned strideElems,
                                                  unsigned padIntervalCode,
                                                  unsigned padAmountCode) {
    unsigned long long ga = (unsigned long long)(uintptr_t)gptr;
    su4 g0 = { 1u,                                   // count=1 (valid user D#)
               lds_off,                              // lds_addr
               (unsigned)ga,                         // global_addr[31:0]
               (unsigned)(ga >> 32) | 0x80000000u }; // global_addr[56:32] | type=2
    su8 g1;
    g1[0] = (1u << 16)                      // data_size = 2 bytes
          | (1u << 20)                      // pad_enable
          | (padIntervalCode << 22)
          | (padAmountCode << 25);
    g1[1] = (rowElems & 0xFFFFu) << 16;     // tensor_dim0[15:0]  (== tile_dim0)
    g1[2] = (rowElems >> 16) | ((rows & 0xFFFFu) << 16);  // dim0 hi | tensor_dim1 lo
    g1[3] = (rows >> 16) | (rowElems << 16);              // dim1 hi | tile_dim0
    g1[4] = rows & 0xFFFFu;                 // tile_dim1 (tile_dim2 = 0)
    g1[5] = strideElems;                    // tensor_dim0_stride[31:0]
    g1[6] = 0u;                             // stride hi | tensor_dim1_stride lo
    g1[7] = 0u;
    asm volatile("tensor_load_to_lds %0, %1" :: "s"(g0), "s"(g1) : "memory");
}

// ---------------------------------------------------------------------------
// Weight repack: Wq/Wk/Wv [NH,D,DK] -> Wqkvt [3072][1024] f16; pack biases.
// ---------------------------------------------------------------------------
__global__ void pack_qkv_w(const float* __restrict__ Wq, const float* __restrict__ Wk,
                           const float* __restrict__ Wv, const float* __restrict__ bq,
                           const float* __restrict__ bk, const float* __restrict__ bv,
                           _Float16* __restrict__ Wt, float* __restrict__ bcat) {
    size_t idx = (size_t)blockIdx.x * 256 + threadIdx.x;   // 3072*1024 elems
    int n = (int)(idx >> 10);
    int d = (int)(idx & 1023);
    int sel = n >> 10;
    int hn  = n & 1023;
    int h   = hn >> 6, kk = hn & 63;
    const float* W = (sel == 0) ? Wq : (sel == 1) ? Wk : Wv;
    Wt[idx] = (_Float16)W[((size_t)h * 1024 + d) * 64 + kk];
    if (d == 0) {
        const float* bb = (sel == 0) ? bq : (sel == 1) ? bk : bv;
        bcat[n] = bb[hn];
    }
}

// Generic transpose + f32->f16: dst[Nc][K], dst[n][k] = src[k][n]
__global__ void transpose_f32_f16(_Float16* __restrict__ dst, const float* __restrict__ src,
                                  int K, int Nc) {
    size_t idx = (size_t)blockIdx.x * 256 + threadIdx.x;
    if (idx >= (size_t)K * Nc) return;
    size_t n = idx / (size_t)K;
    size_t k = idx % (size_t)K;
    dst[idx] = (_Float16)src[k * (size_t)Nc + n];
}

// ---------------------------------------------------------------------------
// Row LayerNorm: f32 [rows][D] -> f16 [rows][D]
// ---------------------------------------------------------------------------
__global__ __launch_bounds__(256) void layernorm_f32_f16(const float* __restrict__ X,
                                                         _Float16* __restrict__ Y, int D) {
    const int row = blockIdx.x, t = threadIdx.x;
    const float* x = X + (size_t)row * D;
    __shared__ float red[256];
    float s = 0.f, ss = 0.f;
    for (int i = t; i < D; i += 256) { float v = x[i]; s += v; ss += v * v; }
    red[t] = s; __syncthreads();
    for (int o = 128; o > 0; o >>= 1) { if (t < o) red[t] += red[t + o]; __syncthreads(); }
    float mean = red[0] / (float)D;
    __syncthreads();
    red[t] = ss; __syncthreads();
    for (int o = 128; o > 0; o >>= 1) { if (t < o) red[t] += red[t + o]; __syncthreads(); }
    float var  = red[0] / (float)D - mean * mean;
    float rstd = rsqrtf(var + 1e-5f);
    _Float16* y = Y + (size_t)row * D;
    for (int i = t; i < D; i += 256) y[i] = (_Float16)((x[i] - mean) * rstd);
}

// ---------------------------------------------------------------------------
// Tiled WMMA GEMM: C[M][Ncols] = A[M][K](f16) * Bt[Ncols][K](f16)^T + bias
//   MODE 0: store f16 (QKV) | 1: +resid f32 (Wo) | 2: ReLU f16 (FFN1)
//   MODE 3: +resid f32 (FFN2 -> out)
// Workgroup: 256 threads (8 waves). Tile 128x128, K-step 32.
// Tiles staged by TDM (wave 0 issues, all waves barrier on completion).
// Wave w computes rows [w*16, w*16+16) x all 128 cols (8 WMMA subtiles).
// ---------------------------------------------------------------------------
template <int MODE>
__global__ __launch_bounds__(256) void gemm_wmma(const _Float16* __restrict__ A,
                                                 const _Float16* __restrict__ Bt,
                                                 const float* __restrict__ bias,
                                                 const float* __restrict__ resid,
                                                 void* __restrict__ outv,
                                                 int M, int Ncols, int K) {
    constexpr int BM = 128, BN = 128, BK = 32, LDT = BK + 8;  // 80 B rows (64B data+16B pad)
    __shared__ __align__(16) _Float16 As[BM * LDT];
    __shared__ __align__(16) _Float16 Bs[BN * LDT];
    const int t = threadIdx.x, w = t >> 5, lane = t & 31;
    const int row0 = blockIdx.y * BM, col0 = blockIdx.x * BN;
    const unsigned asOff = (unsigned)(uintptr_t)(void*)As;
    const unsigned bsOff = (unsigned)(uintptr_t)(void*)Bs;

    v8f acc[8] = {};
    for (int kt = 0; kt < K; kt += BK) {
        __syncthreads();                       // prior compute done before DMA overwrite
        if (w == 0) {
            // pad codes: after 8x8B (=64B row) insert 4 DWORDs (16B) -> LDT=40 halves
            tdm_load_tile_f16(asOff, A  + (size_t)row0 * K + kt, BM, BK, K, 3, 3);
            tdm_load_tile_f16(bsOff, Bt + (size_t)col0 * K + kt, BN, BK, K, 3, 3);
            __builtin_amdgcn_s_wait_tensorcnt(0);
        }
        __syncthreads();                       // tiles visible to all waves
        v16h a = frag_a(As + w * 16 * LDT, LDT, lane, 0);
#pragma unroll
        for (int c = 0; c < 8; ++c) {
            v16h b = frag_b(Bs + c * 16 * LDT, LDT, lane, 0);
            acc[c] = wmma16(a, b, acc[c]);
        }
    }

    // Epilogue. C/D layout: lane -> col = lane%16; vgpr r -> row = (lane/16)*8 + r.
    const int gr0  = row0 + w * 16 + ((lane >> 4) << 3);
    const int nloc = lane & 15;
#pragma unroll
    for (int c = 0; c < 8; ++c) {
        int gc = col0 + c * 16 + nloc;
        float bv = bias ? bias[gc] : 0.f;
#pragma unroll
        for (int r = 0; r < 8; ++r) {
            size_t o = (size_t)(gr0 + r) * Ncols + gc;
            float v = acc[c][r] + bv;
            if constexpr (MODE == 2) v = v > 0.f ? v : 0.f;
            if constexpr (MODE == 1 || MODE == 3) v += resid[o];
            if constexpr (MODE == 0 || MODE == 2)
                ((_Float16*)outv)[o] = (_Float16)v;
            else
                ((float*)outv)[o] = v;
        }
    }
}

// ---------------------------------------------------------------------------
// Flash attention, one workgroup = (batch b, head h, 64 query rows).
// qkv: f16 [B*N][3072] = [Q(1024) | K(1024) | V(1024)], per-head 64 cols.
// scores = Q K^T / 8 + spm; online softmax; O += P V; write O f16 -> Obuf.
// K tile staged by TDM; V loaded+transposed via LDS; Q loaded once.
// ---------------------------------------------------------------------------
__global__ __launch_bounds__(256) void attn_wmma(const _Float16* __restrict__ qkv,
                                                 const float* __restrict__ spm,
                                                 _Float16* __restrict__ Obuf) {
    constexpr int LQ = 72;   // f16 row pitch (144 B = 128B data + 16B pad)
    constexpr int LS = 68;   // f32 row pitch
    __shared__ __align__(16) _Float16 Qs[64 * LQ], Ks[64 * LQ], Vts[64 * LQ], Ps[64 * LQ];
    __shared__ float Sp[64 * LS];
    __shared__ float mrow[64], lrow[64], arow[64];

    const int t = threadIdx.x, w = t >> 5, lane = t & 31;
    const int rb = blockIdx.x & 7;
    const int h  = (blockIdx.x >> 3) & 15;
    const int b  = blockIdx.x >> 7;
    const int row0 = rb * 64;

    const size_t base = ((size_t)b * 512) * 3072 + (size_t)h * 64;
    const _Float16* Qg = qkv + base;
    const _Float16* Kg = qkv + base + 1024;
    const _Float16* Vg = qkv + base + 2048;
    const unsigned ksOff = (unsigned)(uintptr_t)(void*)Ks;

#pragma unroll
    for (int p = 0; p < 2; ++p) {                   // Q tile 64x64
        int e = p * 2048 + t * 8;
        int r = e >> 6, kk = e & 63;
        *(v8h*)&Qs[r * LQ + kk] = *(const v8h*)&Qg[(size_t)(row0 + r) * 3072 + kk];
    }
    if (t < 64) { mrow[t] = -3.0e38f; lrow[t] = 0.f; }

    const int rs = w >> 1;          // row strip 0..3 (16 rows each)
    const int ch = w & 1;           // column half (32 cols each)
    v8f oacc[2] = {};
    const float scale = 0.125f;     // 1/sqrt(64)

    for (int key0 = 0; key0 < 512; key0 += 64) {
        __syncthreads();
        if (w == 0)                                  // K tile [key][dk] via TDM
            tdm_load_tile_f16(ksOff, Kg + (size_t)key0 * 3072, 64, 64, 3072, 4, 3);
#pragma unroll
        for (int p = 0; p < 2; ++p) {                // V^T tile [dk][key]
            int e = p * 2048 + t * 8;
            int j = e >> 6, kk = e & 63;
            v8h vv = *(const v8h*)&Vg[(size_t)(key0 + j) * 3072 + kk];
#pragma unroll
            for (int i = 0; i < 8; ++i) Vts[(kk + i) * LQ + j] = vv[i];
        }
        if (w == 0) __builtin_amdgcn_s_wait_tensorcnt(0);
        __syncthreads();

        // --- scores S = Q K^T (64x64 panel; this wave: 16 rows x 32 cols) ---
        v16h a0 = frag_a(Qs + rs * 16 * LQ, LQ, lane, 0);
        v16h a1 = frag_a(Qs + rs * 16 * LQ, LQ, lane, 32);
#pragma unroll
        for (int sub = 0; sub < 2; ++sub) {
            int cl = ch * 32 + sub * 16;
            v8f s = {};
            s = wmma16(a0, frag_b(Ks + cl * LQ, LQ, lane, 0), s);
            s = wmma16(a1, frag_b(Ks + cl * LQ, LQ, lane, 32), s);
            int lrow0 = rs * 16 + ((lane >> 4) << 3);
            int gcol  = key0 + cl + (lane & 15);
            const float* sp = spm + ((size_t)b * 512 + row0 + lrow0) * 512 + gcol;
#pragma unroll
            for (int r = 0; r < 8; ++r)
                Sp[(lrow0 + r) * LS + cl + (lane & 15)] = s[r] * scale + sp[(size_t)r * 512];
        }
        __syncthreads();

        // --- online softmax bookkeeping: one thread per query row ---
        if (t < 64) {
            float mold = mrow[t], mx = mold;
#pragma unroll 8
            for (int j = 0; j < 64; ++j) { float v = Sp[t * LS + j]; mx = v > mx ? v : mx; }
            float al = __expf(mold - mx), sum = 0.f;
#pragma unroll 8
            for (int j = 0; j < 64; ++j) {
                float p = __expf(Sp[t * LS + j] - mx);
                sum += p;
                Ps[t * LQ + j] = (_Float16)p;
            }
            mrow[t] = mx; lrow[t] = lrow[t] * al + sum; arow[t] = al;
        }
        __syncthreads();

        // --- rescale accumulators and O += P V ---
        {
            int rbase = rs * 16 + ((lane >> 4) << 3);
#pragma unroll
            for (int sub = 0; sub < 2; ++sub)
#pragma unroll
                for (int r = 0; r < 8; ++r) oacc[sub][r] *= arow[rbase + r];
            v16h pa0 = frag_a(Ps + rs * 16 * LQ, LQ, lane, 0);
            v16h pa1 = frag_a(Ps + rs * 16 * LQ, LQ, lane, 32);
#pragma unroll
            for (int sub = 0; sub < 2; ++sub) {
                int cl = ch * 32 + sub * 16;
                oacc[sub] = wmma16(pa0, frag_b(Vts + cl * LQ, LQ, lane, 0), oacc[sub]);
                oacc[sub] = wmma16(pa1, frag_b(Vts + cl * LQ, LQ, lane, 32), oacc[sub]);
            }
        }
    }
    __syncthreads();

    // --- normalize by row sums and write [b,n, h*64+dk] ---
    {
        int rbase = rs * 16 + ((lane >> 4) << 3);
#pragma unroll
        for (int sub = 0; sub < 2; ++sub) {
            int cl = ch * 32 + sub * 16 + (lane & 15);
#pragma unroll
            for (int r = 0; r < 8; ++r) {
                float linv = 1.f / lrow[rbase + r];
                Obuf[((size_t)b * 512 + row0 + rbase + r) * 1024 + h * 64 + cl] =
                    (_Float16)(oacc[sub][r] * linv);
            }
        }
    }
}

// ---------------------------------------------------------------------------
// Host orchestration. Workspace layout (bytes):
//   ln1  f16  [8192][1024]  @ 0            (reused as xn)
//   qkv  f16  [8192][3072]  @ 16,777,216   (reused: hmid spans qkv+O = 64 MB)
//   O    f16  [8192][1024]  @ 67,108,864
//   x    f32  [8192][1024]  @ 83,886,080
//   Wqkvt/Wot/W1t/W2t/bqkv  @ 117,440,512.. (~25.2 MB)   total ~142.6 MB
// ---------------------------------------------------------------------------
extern "C" void kernel_launch(void* const* d_in, const int* in_sizes, int n_in,
                              void* d_out, int out_size, void* d_ws, size_t ws_size,
                              hipStream_t stream) {
    (void)in_sizes; (void)n_in; (void)out_size; (void)ws_size;
    const float* H   = (const float*)d_in[0];
    const float* spm = (const float*)d_in[1];
    const float* Wq  = (const float*)d_in[2];
    const float* bq  = (const float*)d_in[3];
    const float* Wk  = (const float*)d_in[4];
    const float* bk  = (const float*)d_in[5];
    const float* Wv  = (const float*)d_in[6];
    const float* bv  = (const float*)d_in[7];
    const float* Wo  = (const float*)d_in[8];
    const float* bo  = (const float*)d_in[9];
    const float* W1  = (const float*)d_in[10];
    const float* b1  = (const float*)d_in[11];
    const float* W2  = (const float*)d_in[12];
    const float* b2  = (const float*)d_in[13];

    char* ws = (char*)d_ws;
    _Float16* ln1   = (_Float16*)(ws);
    _Float16* qkv   = (_Float16*)(ws + 16777216ull);
    _Float16* Obuf  = (_Float16*)(ws + 67108864ull);
    float*    x     = (float*)   (ws + 83886080ull);
    _Float16* Wqkvt = (_Float16*)(ws + 117440512ull);
    _Float16* Wot   = (_Float16*)(ws + 123731968ull);
    _Float16* W1t   = (_Float16*)(ws + 125829120ull);
    _Float16* W2t   = (_Float16*)(ws + 134217728ull);
    float*    bqkv  = (float*)   (ws + 142606336ull);
    _Float16* hmid  = qkv;   // reuses qkv+Obuf span (exactly 64 MB)
    _Float16* xn    = ln1;   // reuses ln1

    const int M = 16 * 512;  // 8192 token rows

    // 1. Weight repack (f32 -> f16, transposed to [outCol][K])
    pack_qkv_w<<<(3072 * 1024) / 256, 256, 0, stream>>>(Wq, Wk, Wv, bq, bk, bv, Wqkvt, bqkv);
    transpose_f32_f16<<<(1024 * 1024) / 256, 256, 0, stream>>>(Wot, Wo, 1024, 1024);
    transpose_f32_f16<<<(1024 * 4096) / 256, 256, 0, stream>>>(W1t, W1, 1024, 4096);
    transpose_f32_f16<<<(4096 * 1024) / 256, 256, 0, stream>>>(W2t, W2, 4096, 1024);

    // 2. ln1 = LN(H)
    layernorm_f32_f16<<<M, 256, 0, stream>>>(H, ln1, 1024);

    // 3. Fused QKV projection
    gemm_wmma<0><<<dim3(3072 / 128, M / 128), 256, 0, stream>>>(
        ln1, Wqkvt, bqkv, nullptr, qkv, M, 3072, 1024);

    // 4. Attention (flash, per (b,h,64-row tile))
    attn_wmma<<<16 * 16 * 8, 256, 0, stream>>>(qkv, spm, Obuf);

    // 5. x = O @ Wo + bo + H
    gemm_wmma<1><<<dim3(1024 / 128, M / 128), 256, 0, stream>>>(
        Obuf, Wot, bo, H, x, M, 1024, 1024);

    // 6. xn = LN(x)
    layernorm_f32_f16<<<M, 256, 0, stream>>>(x, xn, 1024);

    // 7. h = relu(xn @ W1 + b1)
    gemm_wmma<2><<<dim3(4096 / 128, M / 128), 256, 0, stream>>>(
        xn, W1t, b1, nullptr, hmid, M, 4096, 1024);

    // 8. out = x + h @ W2 + b2
    gemm_wmma<3><<<dim3(1024 / 128, M / 128), 256, 0, stream>>>(
        hmid, W2t, b2, x, (float*)d_out, M, 1024, 4096);
}
